// VisionTransformer_17660905521874
// MI455X (gfx1250) — matrix operand order repaired and checked by reading, not compiled, not run on hardware
//
#include <hip/hip_runtime.h>
#include <hip/hip_bf16.h>
#include <stdint.h>

// ---------------------------------------------------------------------------
// Model constants (fixed by the reference)
// ---------------------------------------------------------------------------
#define NTOK   4096      // B*P
#define SEQ    1024
#define BATCH  4
#define HDIM   512
#define NHEADS 8
#define HD     64
#define FFDIM  2048
#define VOCAB  4096
#define NLAYER 6

typedef __attribute__((ext_vector_type(16))) _Float16 v16h;
typedef __attribute__((ext_vector_type(8)))  _Float16 v8h;
typedef __attribute__((ext_vector_type(8)))  float    v8f;
typedef __attribute__((ext_vector_type(4)))  float    v4f;
typedef __attribute__((ext_vector_type(4)))  unsigned int u32x4;
typedef __attribute__((ext_vector_type(8)))  int      i32x8;
typedef __attribute__((ext_vector_type(4)))  int      i32x4;

union Frag16 { v16h v; v8h h[2]; _Float16 e[16]; };

__device__ __forceinline__ v8f wmma16(const Frag16& a, const Frag16& b, v8f c) {
  return __builtin_amdgcn_wmma_f32_16x16x32_f16(false, a.v, false, b.v,
                                                (short)0, c, false, false);
}

// ---------------------------------------------------------------------------
// TDM: issue a 2D tile load (128 rows x 32 f16) global -> LDS.
// D# built per CDNA5 ISA 8.3/8.4: group0 {count, lds_addr, global_addr, type=2},
// group1 {data_size=2B, tensor dims (large), tile 32x128, dim0 stride = K}.
// This toolchain exposes the 6-arg builtin:
//   (uint32x4 g0, int32x8 g1, int32x4 g2, int32x4 g3, int32x8 extra, i32 cpol)
// ---------------------------------------------------------------------------
__device__ __forceinline__ void tdm_load_tile(uint32_t lds_off, const _Float16* gsrc,
                                              int k_stride_elems) {
  uint64_t ga = (uint64_t)(uintptr_t)gsrc;
  u32x4 g0;
  g0[0] = 1u;                                   // count=1, user descriptor
  g0[1] = lds_off;                              // lds_addr (bytes)
  g0[2] = (uint32_t)ga;                         // global_addr[31:0]
  g0[3] = (uint32_t)(ga >> 32) | (2u << 30);    // global_addr[56:32] | type=2
  i32x8 g1;
  g1[0] = (int)(1u << 16);                      // data_size=1 -> 2 bytes; wg_mask=0
  g1[1] = 0;                                    // atomic_barrier=0; tensor_dim0[15:0]=0
  g1[2] = 0x4000;                               // tensor_dim0 = 0x40000000 (hi16); dim1 lo=0
  g1[3] = 0x4000 | (32 << 16);                  // tensor_dim1 hi16; tile_dim0 = 32
  g1[4] = 128;                                  // tile_dim1 = 128; tile_dim2 = 0
  g1[5] = k_stride_elems;                       // tensor_dim0_stride[31:0]
  g1[6] = 0;                                    // stride[47:32]=0; dim1_stride lo=0
  g1[7] = 0;
  i32x4 gz4; gz4[0] = 0; gz4[1] = 0; gz4[2] = 0; gz4[3] = 0;
  i32x8 gz8;
  #pragma unroll
  for (int t = 0; t < 8; t++) gz8[t] = 0;
#if __has_builtin(__builtin_amdgcn_tensor_load_to_lds)
  __builtin_amdgcn_tensor_load_to_lds(g0, g1, gz4, gz4, gz8, 0);
#endif
}

// ---------------------------------------------------------------------------
// Weight conversion kernels
// ---------------------------------------------------------------------------
__global__ __launch_bounds__(256) void cvt_f16_kernel(const float* __restrict__ src,
                                                      _Float16* __restrict__ dst, int n) {
  int i = blockIdx.x * 256 + threadIdx.x;
  if (i < n) dst[i] = (_Float16)src[i];
}

// emb_w [512,16] -> f16 padded to [512,32] (K padded to 32 for WMMA)
__global__ __launch_bounds__(256) void emb_pad_kernel(const float* __restrict__ emb,
                                                      _Float16* __restrict__ dst) {
  int i = blockIdx.x * 256 + threadIdx.x;   // 512*32
  int r = i >> 5, c = i & 31;
  dst[i] = (c < 16) ? (_Float16)emb[r * 16 + c] : (_Float16)0.0f;
}

// relb [L,65,NH] -> relm [L,65] (mean over heads)
__global__ void relmean_kernel(const float* __restrict__ relb, float* __restrict__ relm) {
  int i = blockIdx.x * 64 + threadIdx.x;
  if (i < NLAYER * 65) {
    float s = 0.f;
    #pragma unroll
    for (int h = 0; h < NHEADS; h++) s += relb[i * NHEADS + h];
    relm[i] = s * (1.0f / NHEADS);
  }
}

// ---------------------------------------------------------------------------
// Tokenize: one wave32 per patch. Emits f16 proc (K padded to 32) directly.
// ---------------------------------------------------------------------------
__global__ __launch_bounds__(256)
void tokenize_kernel(const float* __restrict__ patches, const float* __restrict__ vocab,
                     const float* __restrict__ masked_tok, const float* __restrict__ partial_p,
                     const int* __restrict__ partial_tab, const int* __restrict__ fully_idx_p,
                     _Float16* __restrict__ proc16, float* __restrict__ tok_out) {
  const int patch = blockIdx.x * 8 + (threadIdx.x >> 5);
  const int lane  = threadIdx.x & 31;
  const float* p = patches + (size_t)patch * 16;

  float pv[16];
  #pragma unroll
  for (int d = 0; d < 16; d++) pv[d] = p[d];
  float myp = 0.f;
  #pragma unroll
  for (int d = 0; d < 16; d++) if (lane == d) myp = pv[d];

  bool mymask = (lane < 16) && (fabsf(myp - 0.5f) < 0.1f);
  unsigned mb   = (unsigned)__ballot(mymask) & 0xFFFFu;
  bool fully    = (mb == 0xFFFFu);
  bool hasmask  = (mb != 0u);
  unsigned notm = (~mb) & 0xFFFFu;
  int first_bin = notm ? (__ffs((int)notm) - 1) : 0;
  float fv  = __shfl(myp, first_bin);
  int  val  = (int)rintf(fv);
  int  pidx = first_bin * 2 + val;
  pidx = pidx < 0 ? 0 : (pidx > 31 ? 31 : pidx);
  bool partial = hasmask && !fully;

  // nearest vocab token: argmin ||v||^2 - 2 p.v
  float bestd = 3.0e38f; int besti = 0;
  for (int v0 = lane; v0 < VOCAB; v0 += 32) {
    const float* vv = vocab + (size_t)v0 * 16;
    float nrm = 0.f, dot = 0.f;
    #pragma unroll
    for (int d = 0; d < 16; d++) { float w = vv[d]; nrm += w * w; dot += w * pv[d]; }
    float dist = nrm - 2.0f * dot;
    if (dist < bestd) { bestd = dist; besti = v0; }
  }
  #pragma unroll
  for (int o = 16; o > 0; o >>= 1) {
    float od = __shfl_xor(bestd, o);
    int   oi = __shfl_xor(besti, o);
    if (od < bestd || (od == bestd && oi < besti)) { bestd = od; besti = oi; }
  }

  int tok = fully ? fully_idx_p[0] : (partial ? partial_tab[pidx] : besti);
  if (lane == 0) tok_out[patch] = (float)tok;

  if (lane < 16) {
    float pr = fully ? masked_tok[lane]
                     : (partial ? partial_p[pidx * 16 + lane] : myp);
    proc16[(size_t)patch * 32 + lane] = (_Float16)pr;
  } else {
    proc16[(size_t)patch * 32 + lane] = (_Float16)0.0f;  // K padding 16..31
  }
}

// ---------------------------------------------------------------------------
// LayerNorm: one wave per row of H=512
// ---------------------------------------------------------------------------
__global__ __launch_bounds__(256)
void layernorm_kernel(const float* __restrict__ x, const float* __restrict__ w,
                      const float* __restrict__ b, float* __restrict__ y) {
  const int row  = blockIdx.x * 8 + (threadIdx.x >> 5);
  const int lane = threadIdx.x & 31;
  const float* xr = x + (size_t)row * HDIM;
  float v[16]; float s = 0.f;
  #pragma unroll
  for (int j = 0; j < 16; j++) { v[j] = xr[lane + j * 32]; s += v[j]; }
  #pragma unroll
  for (int o = 16; o > 0; o >>= 1) s += __shfl_xor(s, o);
  float mu = s * (1.0f / HDIM);
  float q = 0.f;
  #pragma unroll
  for (int j = 0; j < 16; j++) { float d = v[j] - mu; q += d * d; }
  #pragma unroll
  for (int o = 16; o > 0; o >>= 1) q += __shfl_xor(q, o);
  float rstd = rsqrtf(q * (1.0f / HDIM) + 1e-5f);
  #pragma unroll
  for (int j = 0; j < 16; j++) {
    int c = lane + j * 32;
    y[(size_t)row * HDIM + c] = (v[j] - mu) * rstd * w[c] + b[c];
  }
}

// ---------------------------------------------------------------------------
// WMMA GEMM with TDM tile feed, double-buffered.
// D[M,N] = act(A[M,K](f16) @ W[N,K](f16)^T + bias + residual)
// Block tile 128x128, K-step 32, 256 threads = 8 waves (4 row x 2 col slices).
// Wave 0 issues TENSOR_LOAD_TO_LDS for the next A/B tile pair while all waves
// run WMMA on the current pair; s_wait_tensorcnt(0) gates buffer reuse.
// ---------------------------------------------------------------------------
__global__ __launch_bounds__(256)
void gemm_wmma_kernel(const _Float16* __restrict__ A, const _Float16* __restrict__ W,
                      const float* __restrict__ bias, const float* __restrict__ res,
                      float* __restrict__ D, int M, int N, int K, int act) {
  __shared__ _Float16 sA[2][128][32];
  __shared__ _Float16 sB[2][128][32];
  const int tid  = threadIdx.x;
  const int lane = tid & 31;
  const int wave = tid >> 5;
  const int l16  = lane & 15;
  const int hf   = lane >> 4;
  const int wm   = wave & 3;   // 32-row slice
  const int wn   = wave >> 2;  // 64-col slice
  const int brow = blockIdx.y * 128;
  const int bcol = blockIdx.x * 128;

  v8f acc[2][4];
  #pragma unroll
  for (int i = 0; i < 2; i++)
    #pragma unroll
    for (int j = 0; j < 4; j++)
      #pragma unroll
      for (int e = 0; e < 8; e++) acc[i][j][e] = 0.0f;

  const bool issuer = (wave == 0);
  if (issuer) {
    tdm_load_tile((uint32_t)(uintptr_t)&sA[0][0][0], A + (size_t)brow * K, K);
    tdm_load_tile((uint32_t)(uintptr_t)&sB[0][0][0], W + (size_t)bcol * K, K);
  }

  const int nk = K >> 5;
  for (int ki = 0; ki < nk; ki++) {
    const int buf = ki & 1;
    if (issuer) __builtin_amdgcn_s_wait_tensorcnt(0);  // current pair landed
    __syncthreads();
    if (issuer && (ki + 1) < nk) {                     // prefetch next pair
      const int k0 = (ki + 1) << 5;
      tdm_load_tile((uint32_t)(uintptr_t)&sA[buf ^ 1][0][0], A + (size_t)brow * K + k0, K);
      tdm_load_tile((uint32_t)(uintptr_t)&sB[buf ^ 1][0][0], W + (size_t)bcol * K + k0, K);
    }

    // ---- fragments + wmma ----
    Frag16 af[2], bf[4];
    #pragma unroll
    for (int i = 0; i < 2; i++) {
      const _Float16* p = &sA[buf][wm * 32 + i * 16 + l16][0];
      af[i].h[0] = *(const v8h*)(p + hf * 8);
      af[i].h[1] = *(const v8h*)(p + 16 + hf * 8);
    }
    #pragma unroll
    for (int j = 0; j < 4; j++) {
      const _Float16* p = &sB[buf][wn * 64 + j * 16 + l16][0];
      bf[j].h[0] = *(const v8h*)(p + hf * 8);
      bf[j].h[1] = *(const v8h*)(p + 16 + hf * 8);
    }
    #pragma unroll
    for (int i = 0; i < 2; i++)
      #pragma unroll
      for (int j = 0; j < 4; j++)
        acc[i][j] = wmma16(af[i], bf[j], acc[i][j]);
    __syncthreads();
  }

  // ---- epilogue ----
  #pragma unroll
  for (int i = 0; i < 2; i++) {
    #pragma unroll
    for (int j = 0; j < 4; j++) {
      const int col = bcol + wn * 64 + j * 16 + l16;
      float bv = bias ? bias[col] : 0.0f;
      #pragma unroll
      for (int e = 0; e < 8; e++) {
        const int row = brow + wm * 32 + i * 16 + e + 8 * hf;
        float v = acc[i][j][e] + bv;
        if (res) v += res[(size_t)row * N + col];
        if (act == 1) v = 0.5f * v * (1.0f + erff(v * 0.70710678118654752f));
        D[(size_t)row * N + col] = v;
      }
    }
  }
}

// ---------------------------------------------------------------------------
// Flash attention with WMMA.
// grid = (SEQ/128, NHEADS, BATCH); 256 threads = 8 waves, each wave a 16-row
// query tile. K/V 32x64 tiles staged in LDS (f32->f16) cooperatively.
// ---------------------------------------------------------------------------
__global__ __launch_bounds__(256)
void attention_kernel(const float* __restrict__ qkv, const float* __restrict__ relm,
                      float* __restrict__ out) {
  __shared__ _Float16 sK[32][64];
  __shared__ _Float16 sV[32][64];
  __shared__ _Float16 sP[8][16][32];
  __shared__ float    sRel[65];

  const int tid  = threadIdx.x, lane = tid & 31, wave = tid >> 5;
  const int l16  = lane & 15, hf = lane >> 4;
  const int b    = blockIdx.z, h = blockIdx.y;
  const int qbase = blockIdx.x * 128 + wave * 16;   // within-sequence
  const size_t tokbase = (size_t)b * SEQ;

  if (tid < 65) sRel[tid] = relm[tid];

  // Q fragments for k-chunks 0..31, 32..63 (scaled by 1/sqrt(hd) = 0.125)
  Frag16 qf[2];
  {
    const float* qp = qkv + (tokbase + qbase + l16) * (3 * HDIM) + h * HD;
    #pragma unroll
    for (int c = 0; c < 2; c++)
      #pragma unroll
      for (int j = 0; j < 8; j++) {
        qf[c].e[j]     = (_Float16)(qp[c * 32 + hf * 8 + j] * 0.125f);
        qf[c].e[j + 8] = (_Float16)(qp[c * 32 + 16 + hf * 8 + j] * 0.125f);
      }
  }

  float mrun[8], lrun[8];
  v8f o[4];
  #pragma unroll
  for (int e = 0; e < 8; e++) { mrun[e] = -3.0e38f; lrun[e] = 0.0f; }
  #pragma unroll
  for (int j = 0; j < 4; j++)
    #pragma unroll
    for (int e = 0; e < 8; e++) o[j][e] = 0.0f;

  for (int kb = 0; kb < SEQ; kb += 32) {
    __syncthreads();
    {   // stage 32 keys x 64 dims of K and V
      int r  = tid >> 3;
      int c0 = (tid & 7) * 8;
      const float* kp = qkv + (tokbase + kb + r) * (3 * HDIM) + HDIM + h * HD + c0;
      const float* vp = qkv + (tokbase + kb + r) * (3 * HDIM) + 2 * HDIM + h * HD + c0;
      v8h hk, hv;
      #pragma unroll
      for (int t = 0; t < 8; t++) { hk[t] = (_Float16)kp[t]; hv[t] = (_Float16)vp[t]; }
      *(v8h*)&sK[r][c0] = hk;
      *(v8h*)&sV[r][c0] = hv;
    }
    __syncthreads();

    // scores: two 16x16 tiles (keys kb..kb+15, kb+16..kb+31)
    v8f s[2];
    #pragma unroll
    for (int kt = 0; kt < 2; kt++) {
      Frag16 bfr[2];
      const int keyl = kt * 16 + l16;
      #pragma unroll
      for (int c = 0; c < 2; c++) {
        const _Float16* p = &sK[keyl][c * 32 + hf * 8];
        bfr[c].h[0] = *(const v8h*)p;
        bfr[c].h[1] = *(const v8h*)(p + 16);
      }
      v8f sc;
      #pragma unroll
      for (int e = 0; e < 8; e++) sc[e] = 0.0f;
      sc = wmma16(qf[0], bfr[0], sc);
      sc = wmma16(qf[1], bfr[1], sc);
      s[kt] = sc;
    }

    // add head-averaged relative bias
    #pragma unroll
    for (int kt = 0; kt < 2; kt++)
      #pragma unroll
      for (int e = 0; e < 8; e++) {
        int qr = qbase + e + 8 * hf;
        int kr = kb + kt * 16 + l16;
        int rel = qr - kr;
        rel = rel < -32 ? -32 : (rel > 32 ? 32 : rel);
        s[kt][e] += sRel[rel + 32];
      }

    // online softmax
    float scal[8];
    #pragma unroll
    for (int e = 0; e < 8; e++) {
      float mx = fmaxf(s[0][e], s[1][e]);
      #pragma unroll
      for (int off = 1; off < 16; off <<= 1) mx = fmaxf(mx, __shfl_xor(mx, off));
      float mn = fmaxf(mrun[e], mx);
      scal[e]  = __expf(mrun[e] - mn);
      s[0][e]  = __expf(s[0][e] - mn);
      s[1][e]  = __expf(s[1][e] - mn);
      float rs = s[0][e] + s[1][e];
      #pragma unroll
      for (int off = 1; off < 16; off <<= 1) rs += __shfl_xor(rs, off);
      lrun[e] = lrun[e] * scal[e] + rs;
      mrun[e] = mn;
    }
    #pragma unroll
    for (int j = 0; j < 4; j++)
      #pragma unroll
      for (int e = 0; e < 8; e++) o[j][e] *= scal[e];

    // re-layout P (C-layout) -> A-fragment via per-wave LDS round-trip
    #pragma unroll
    for (int kt = 0; kt < 2; kt++)
      #pragma unroll
      for (int e = 0; e < 8; e++)
        sP[wave][e + 8 * hf][kt * 16 + l16] = (_Float16)s[kt][e];
    asm volatile("s_wait_dscnt 0" ::: "memory");   // same-wave LDS RAW fence

    Frag16 pf;
    {
      const _Float16* pp = &sP[wave][l16][0];
      pf.h[0] = *(const v8h*)(pp + hf * 8);
      pf.h[1] = *(const v8h*)(pp + 16 + hf * 8);
    }
    #pragma unroll
    for (int j = 0; j < 4; j++) {
      Frag16 vf;
      #pragma unroll
      for (int jj = 0; jj < 16; jj++) vf.e[jj] = sV[hf * 8 + jj + (jj >= 8 ? 8 : 0)][j * 16 + l16];
      o[j] = wmma16(pf, vf, o[j]);
    }
  }

  // epilogue: O /= l, scatter to [token][HDIM] at head slice
  #pragma unroll
  for (int j = 0; j < 4; j++)
    #pragma unroll
    for (int e = 0; e < 8; e++) {
      const int row = qbase + e + 8 * hf;
      out[(tokbase + row) * HDIM + h * HD + j * 16 + l16] = o[j][e] / lrun[e];
    }
}

// ---------------------------------------------------------------------------
// Host-side orchestration
// ---------------------------------------------------------------------------
extern "C" void kernel_launch(void* const* d_in, const int* in_sizes, int n_in,
                              void* d_out, int out_size, void* d_ws, size_t ws_size,
                              hipStream_t stream) {
  (void)in_sizes; (void)n_in; (void)out_size; (void)ws_size;
  const float* patches    = (const float*)d_in[0];
  const float* vocab      = (const float*)d_in[1];
  const float* masked_tok = (const float*)d_in[2];
  const float* partial_p  = (const float*)d_in[3];
  const int*   partial_t  = (const int*)d_in[4];
  const int*   fully_idx  = (const int*)d_in[5];
  const float* emb_w  = (const float*)d_in[6];
  const float* emb_b  = (const float*)d_in[7];
  const float* ln1_w  = (const float*)d_in[8];
  const float* ln1_b  = (const float*)d_in[9];
  const float* qkv_w  = (const float*)d_in[10];
  const float* qkv_b  = (const float*)d_in[11];
  const float* out_w  = (const float*)d_in[12];
  const float* out_b  = (const float*)d_in[13];
  const float* relb   = (const float*)d_in[14];
  const float* ln2_w  = (const float*)d_in[15];
  const float* ln2_b  = (const float*)d_in[16];
  const float* ffn_w1 = (const float*)d_in[17];
  const float* ffn_b1 = (const float*)d_in[18];
  const float* ffn_w2 = (const float*)d_in[19];
  const float* ffn_b2 = (const float*)d_in[20];
  const float* proj_w = (const float*)d_in[21];
  const float* proj_b = (const float*)d_in[22];

  float* logits = (float*)d_out;                       // [NTOK, VOCAB]
  float* tokout = logits + (size_t)NTOK * VOCAB;       // [NTOK] as float

  // Alias big scratch into the logits region (overwritten only by final GEMM)
  float* qkvb = logits;                                  // [NTOK,1536]
  float* ffh  = logits + (size_t)NTOK * (3 * HDIM);      // [NTOK,FFDIM]

  // Workspace carve
  uintptr_t base = (uintptr_t)d_ws;
  auto carve = [&](size_t bytes) -> void* {
    uintptr_t p = (base + 255) & ~(uintptr_t)255;
    base = p + bytes;
    return (void*)p;
  };
  _Float16* h_emb  = (_Float16*)carve((size_t)512 * 32 * 2);
  _Float16* h_qkv  = (_Float16*)carve((size_t)NLAYER * 3 * HDIM * HDIM * 2);
  _Float16* h_out  = (_Float16*)carve((size_t)NLAYER * HDIM * HDIM * 2);
  _Float16* h_ff1  = (_Float16*)carve((size_t)NLAYER * FFDIM * HDIM * 2);
  _Float16* h_ff2  = (_Float16*)carve((size_t)NLAYER * HDIM * FFDIM * 2);
  _Float16* h_proj = (_Float16*)carve((size_t)VOCAB * HDIM * 2);
  _Float16* proc16 = (_Float16*)carve((size_t)NTOK * 32 * 2);
  _Float16* a16    = (_Float16*)carve((size_t)NTOK * FFDIM * 2);  // shared A16 scratch
  float* x    = (float*)carve((size_t)NTOK * HDIM * 4);
  float* xn   = (float*)carve((size_t)NTOK * HDIM * 4);
  float* attn = (float*)carve((size_t)NTOK * HDIM * 4);
  float* relm = (float*)carve((size_t)NLAYER * 65 * 4);

  auto cvt = [&](const float* s, _Float16* d, size_t n) {
    cvt_f16_kernel<<<(unsigned)((n + 255) / 256), 256, 0, stream>>>(s, d, (int)n);
  };

  // Weight conversion to f16
  emb_pad_kernel<<<512 * 32 / 256, 256, 0, stream>>>(emb_w, h_emb);
  cvt(qkv_w,  h_qkv,  (size_t)NLAYER * 3 * HDIM * HDIM);
  cvt(out_w,  h_out,  (size_t)NLAYER * HDIM * HDIM);
  cvt(ffn_w1, h_ff1,  (size_t)NLAYER * FFDIM * HDIM);
  cvt(ffn_w2, h_ff2,  (size_t)NLAYER * HDIM * FFDIM);
  cvt(proj_w, h_proj, (size_t)VOCAB * HDIM);
  relmean_kernel<<<(NLAYER * 65 + 63) / 64, 64, 0, stream>>>(relb, relm);

  // Tokenize (emits f16 proc) + embed
  tokenize_kernel<<<NTOK / 8, 256, 0, stream>>>(patches, vocab, masked_tok,
                                                partial_p, partial_t, fully_idx,
                                                proc16, tokout);
  gemm_wmma_kernel<<<dim3(HDIM / 128, NTOK / 128), 256, 0, stream>>>(
      proc16, h_emb, emb_b, nullptr, x, NTOK, HDIM, 32, 0);

  for (int l = 0; l < NLAYER; l++) {
    layernorm_kernel<<<NTOK / 8, 256, 0, stream>>>(x, ln1_w + l * HDIM, ln1_b + l * HDIM, xn);
    cvt(xn, a16, (size_t)NTOK * HDIM);
    gemm_wmma_kernel<<<dim3(3 * HDIM / 128, NTOK / 128), 256, 0, stream>>>(
        a16, h_qkv + (size_t)l * 3 * HDIM * HDIM, qkv_b + (size_t)l * 3 * HDIM,
        nullptr, qkvb, NTOK, 3 * HDIM, HDIM, 0);
    attention_kernel<<<dim3(SEQ / 128, NHEADS, BATCH), 256, 0, stream>>>(
        qkvb, relm + l * 65, attn);
    cvt(attn, a16, (size_t)NTOK * HDIM);
    gemm_wmma_kernel<<<dim3(HDIM / 128, NTOK / 128), 256, 0, stream>>>(
        a16, h_out + (size_t)l * HDIM * HDIM, out_b + (size_t)l * HDIM,
        x, x, NTOK, HDIM, HDIM, 0);                       // + residual
    layernorm_kernel<<<NTOK / 8, 256, 0, stream>>>(x, ln2_w + l * HDIM, ln2_b + l * HDIM, xn);
    cvt(xn, a16, (size_t)NTOK * HDIM);
    gemm_wmma_kernel<<<dim3(FFDIM / 128, NTOK / 128), 256, 0, stream>>>(
        a16, h_ff1 + (size_t)l * FFDIM * HDIM, ffn_b1 + (size_t)l * FFDIM,
        nullptr, ffh, NTOK, FFDIM, HDIM, 1);              // GELU
    cvt(ffh, a16, (size_t)NTOK * FFDIM);
    gemm_wmma_kernel<<<dim3(HDIM / 128, NTOK / 128), 256, 0, stream>>>(
        a16, h_ff2 + (size_t)l * HDIM * FFDIM, ffn_b2 + (size_t)l * HDIM,
        x, x, NTOK, HDIM, FFDIM, 0);                      // + residual
  }

  // Final vocab projection straight into d_out (overwrites aliased scratch)
  cvt(x, a16, (size_t)NTOK * HDIM);
  gemm_wmma_kernel<<<dim3(VOCAB / 128, NTOK / 128), 256, 0, stream>>>(
      a16, h_proj, proj_b, nullptr, logits, NTOK, VOCAB, HDIM, 0);
}
